// SpatioTemporalIntegrator_19224273616912
// MI455X (gfx1250) — compile-verified, matrix operand order checked by reading, not run on hardware
//
#include <hip/hip_runtime.h>
#include <hip/hip_bf16.h>
#include <math.h>

#define DIMD 256
#define HIDD 1024
#define BATCH 2
#define SEQ 4096
#define MTOK (BATCH*SEQ)   // 8192 tokens

typedef __attribute__((ext_vector_type(16))) __bf16 v16bf;
typedef __attribute__((ext_vector_type(8)))  float  v8f;

__device__ __forceinline__ unsigned short f2bfu(float x) {
  union { float f; unsigned u; } a; a.f = x;
  return (unsigned short)((a.u + 0x7FFFu + ((a.u >> 16) & 1u)) >> 16); // RNE
}

union FragBF { v16bf v; uint4 q[2]; };

// ---------------------------------------------------------------------------
// C[M,N] = act( scale * (A[M,K](bf16) @ W[N,K](bf16)^T) + bias )  -> epilogue
// outmode: 0 = fp32 (optional +=), 1 = bf16, 2 = bf16 transposed [N,M],
//          3 = fp32 final: out = x * epT + epS
// Block: 256 threads = 8 waves (2M x 4N); wave tile 64x32 (4x2 frags);
// block tile 128x128, K-step 32.
// Staging: GLOBAL_LOAD_ASYNC_TO_LDS_B128 (ASYNCcnt) + double-buffered LDS.
// ---------------------------------------------------------------------------
__global__ __launch_bounds__(256)
void gemm_wmma_bf16(const unsigned short* __restrict__ A, int lda,
                    const unsigned short* __restrict__ W, int ldw,
                    const float* __restrict__ bias,
                    float scale, int act, int K,
                    int outmode, int accum,
                    void* __restrict__ outp, int ldc,
                    const float* __restrict__ epT,
                    const float* __restrict__ epS)
{
  constexpr int BM = 128, BN = 128, BK = 32, LS = 56; // 112B pitch: 16B aligned, conflict-free
  constexpr int MI = 4, NJ = 2;                       // frags per wave
  __shared__ alignas(16) unsigned short As[2][BM * LS];
  __shared__ alignas(16) unsigned short Ws[2][BN * LS];

  const int tid    = threadIdx.x;
  const int lane   = tid & 31;
  const int wid    = tid >> 5;       // 0..7
  const int wm     = wid >> 2;       // 0..1 (M direction, 64 rows each)
  const int wn     = wid & 3;        // 0..3 (N direction, 32 cols each)
  const int blockM = blockIdx.y * BM;
  const int blockN = blockIdx.x * BN;

  v8f acc[MI][NJ] = {};

  const int lrow  = lane & 15;
  const int khalf = (lane >> 4) << 3;   // 0 or 8: WMMA 16-bit A/B K-split

  // Async memory->LDS staging of one K-tile into buffer `buf`.
  auto stage_async = [&](int buf, int k0) {
    #pragma unroll
    for (int c = tid; c < (BM * BK) / 8; c += 256) {       // A: 512 16B chunks
      const int r = c >> 2, col = (c & 3) << 3;
      const unsigned ldsa = (unsigned)(uintptr_t)&As[buf][r * LS + col];
      const unsigned short* g = &A[(size_t)(blockM + r) * lda + (k0 + col)];
      asm volatile("global_load_async_to_lds_b128 %0, %1, off"
                   :: "v"(ldsa), "v"(g) : "memory");
    }
    #pragma unroll
    for (int c = tid; c < (BN * BK) / 8; c += 256) {       // W: 512 16B chunks
      const int r = c >> 2, col = (c & 3) << 3;
      const unsigned ldsw = (unsigned)(uintptr_t)&Ws[buf][r * LS + col];
      const unsigned short* g = &W[(size_t)(blockN + r) * ldw + (k0 + col)];
      asm volatile("global_load_async_to_lds_b128 %0, %1, off"
                   :: "v"(ldsw), "v"(g) : "memory");
    }
  };

  // Prologue: stage first tile, wait, sync.
  stage_async(0, 0);
  asm volatile("s_wait_asynccnt 0" ::: "memory");
  __syncthreads();

  for (int k0 = 0; k0 < K; k0 += BK) {
    const int buf = (k0 / BK) & 1;

    // Kick off DMA of next tile into the other buffer (overlaps the WMMAs).
    if (k0 + BK < K) stage_async(buf ^ 1, k0 + BK);
    // Warm L2 two tiles ahead.
    if (k0 + 2 * BK < K) {
      const int r = tid >> 1;
      __builtin_prefetch(&A[(size_t)(blockM + r) * lda + (k0 + 2 * BK) + (tid & 1) * 16], 0, 1);
      __builtin_prefetch(&W[(size_t)(blockN + r) * ldw + (k0 + 2 * BK) + (tid & 1) * 16], 0, 1);
    }

    FragBF af[MI], bfr[NJ];
    #pragma unroll
    for (int i = 0; i < MI; ++i) {
      const int r = wm * 64 + i * 16 + lrow;
      af[i].q[0] = *(const uint4*)&As[buf][r * LS + khalf];
      af[i].q[1] = *(const uint4*)&As[buf][r * LS + khalf + 16];
    }
    #pragma unroll
    for (int j = 0; j < NJ; ++j) {
      const int r = wn * 32 + j * 16 + lrow;
      bfr[j].q[0] = *(const uint4*)&Ws[buf][r * LS + khalf];
      bfr[j].q[1] = *(const uint4*)&Ws[buf][r * LS + khalf + 16];
    }
    #pragma unroll
    for (int i = 0; i < MI; ++i)
      #pragma unroll
      for (int j = 0; j < NJ; ++j)
        acc[i][j] = __builtin_amdgcn_wmma_f32_16x16x32_bf16(
            false, af[i].v, false, bfr[j].v, (short)0, acc[i][j], false, false);

    // Our async copies must land before anyone reads buf^1 next iteration.
    asm volatile("s_wait_asynccnt 0" ::: "memory");
    __syncthreads();
  }

  // Epilogue. C/D layout: lane n = lane&15; rows = r + (lane>=16 ? 8 : 0)
  const int rbase = (lane >> 4) << 3;
  #pragma unroll
  for (int i = 0; i < MI; ++i) {
    #pragma unroll
    for (int j = 0; j < NJ; ++j) {
      const int n = blockN + wn * 32 + j * 16 + lrow;
      const float bv = bias ? bias[n] : 0.0f;
      #pragma unroll
      for (int r = 0; r < 8; ++r) {
        const int m = blockM + wm * 64 + i * 16 + rbase + r;
        float x = acc[i][j][r] * scale + bv;
        if (act == 1) x = 0.5f * x * (1.0f + erff(x * 0.70710678118654752f)); // exact gelu
        if (outmode == 0) {
          float* O = (float*)outp;
          const size_t idx = (size_t)m * ldc + n;
          if (accum) O[idx] += x; else O[idx] = x;
        } else if (outmode == 1) {
          ((unsigned short*)outp)[(size_t)m * ldc + n] = f2bfu(x);
        } else if (outmode == 2) {
          ((unsigned short*)outp)[(size_t)n * ldc + m] = f2bfu(x);   // V^T store
        } else {
          float* O = (float*)outp;
          const size_t idx = (size_t)m * ldc + n;
          O[idx] = x * epT[idx] + epS[idx];                          // combined_fc*T + S
        }
      }
    }
  }
}

// ---------------------------------------------------------------------------
// Row softmax over 4096 fp32 cols -> bf16 probs. One block per row.
// ---------------------------------------------------------------------------
__global__ __launch_bounds__(256)
void softmax_rows_4096(const float* __restrict__ sc, unsigned short* __restrict__ p)
{
  const int row = blockIdx.x, tid = threadIdx.x;
  const float* r = sc + (size_t)row * SEQ;
  float x[16];
  float mx = -3.4e38f;
  #pragma unroll
  for (int i = 0; i < 16; ++i) { x[i] = r[tid + i * 256]; mx = fmaxf(mx, x[i]); }
  #pragma unroll
  for (int off = 16; off; off >>= 1) mx = fmaxf(mx, __shfl_xor(mx, off, 32));
  __shared__ float redm[8], reds[8];
  const int wid = tid >> 5, lane = tid & 31;
  if (lane == 0) redm[wid] = mx;
  __syncthreads();
  mx = redm[0];
  #pragma unroll
  for (int i = 1; i < 8; ++i) mx = fmaxf(mx, redm[i]);
  float s = 0.0f;
  #pragma unroll
  for (int i = 0; i < 16; ++i) { x[i] = __expf(x[i] - mx); s += x[i]; }
  #pragma unroll
  for (int off = 16; off; off >>= 1) s += __shfl_xor(s, off, 32);
  if (lane == 0) reds[wid] = s;
  __syncthreads();
  s = reds[0] + reds[1] + reds[2] + reds[3] + reds[4] + reds[5] + reds[6] + reds[7];
  const float inv = 1.0f / s;
  unsigned short* pr = p + (size_t)row * SEQ;
  #pragma unroll
  for (int i = 0; i < 16; ++i) pr[tid + i * 256] = f2bfu(x[i] * inv);
}

__global__ void f32_to_bf16_k(const float* __restrict__ in,
                              unsigned short* __restrict__ out, int n)
{
  int i = blockIdx.x * 256 + threadIdx.x;
  if (i < n) out[i] = f2bfu(in[i]);
}

// ---------------------------------------------------------------------------
extern "C" void kernel_launch(void* const* d_in, const int* in_sizes, int n_in,
                              void* d_out, int out_size, void* d_ws, size_t ws_size,
                              hipStream_t stream)
{
  (void)in_sizes; (void)n_in; (void)out_size; (void)ws_size;
  const float* S   = (const float*)d_in[0];
  const float* T   = (const float*)d_in[1];
  const float* b1  = (const float*)d_in[3];
  const float* b2  = (const float*)d_in[5];
  const float* fcb = (const float*)d_in[7];

  char* wsb = (char*)d_ws;
  size_t off = 0;
  auto alloc = [&](size_t bytes) -> char* {
    char* p = wsb + off; off += (bytes + 255) & ~((size_t)255); return p;
  };

  unsigned short* Sb    = (unsigned short*)alloc((size_t)MTOK * DIMD * 2);
  unsigned short* Tb    = (unsigned short*)alloc((size_t)MTOK * DIMD * 2);
  unsigned short* w1b   = (unsigned short*)alloc((size_t)HIDD * DIMD * 2);
  unsigned short* w2b   = (unsigned short*)alloc((size_t)DIMD * HIDD * 2);
  unsigned short* fcwb  = (unsigned short*)alloc((size_t)DIMD * DIMD * 2);
  unsigned short* awb[12];
  for (int i = 0; i < 12; ++i) awb[i] = (unsigned short*)alloc((size_t)DIMD * DIMD * 2);
  unsigned short* Hb    = (unsigned short*)alloc((size_t)MTOK * HIDD * 2);
  unsigned short* Smb   = (unsigned short*)alloc((size_t)MTOK * DIMD * 2);
  unsigned short* Tmb   = (unsigned short*)alloc((size_t)MTOK * DIMD * 2);
  unsigned short* Qb    = (unsigned short*)alloc((size_t)MTOK * DIMD * 2);
  unsigned short* Kbuf  = (unsigned short*)alloc((size_t)MTOK * DIMD * 2);
  unsigned short* Vt    = (unsigned short*)alloc((size_t)DIMD * MTOK * 2);
  float*          Sc    = (float*)alloc((size_t)SEQ * SEQ * 4);
  unsigned short* Pb    = (unsigned short*)alloc((size_t)SEQ * SEQ * 2);
  float*          Comb  = (float*)alloc((size_t)MTOK * DIMD * 4);
  unsigned short* Combb = (unsigned short*)alloc((size_t)MTOK * DIMD * 2);

  auto cvt = [&](const float* src, unsigned short* dst, int n) {
    f32_to_bf16_k<<<(n + 255) / 256, 256, 0, stream>>>(src, dst, n);
  };
  cvt(S, Sb, MTOK * DIMD);
  cvt(T, Tb, MTOK * DIMD);
  cvt((const float*)d_in[2], w1b,  HIDD * DIMD);
  cvt((const float*)d_in[4], w2b,  DIMD * HIDD);
  cvt((const float*)d_in[6], fcwb, DIMD * DIMD);
  for (int i = 0; i < 12; ++i) cvt((const float*)d_in[8 + 2 * i], awb[i], DIMD * DIMD);

  auto gemm = [&](const unsigned short* A, int lda, const unsigned short* W, int ldw,
                  const float* bias, float scale, int act, int M, int N, int K,
                  int outmode, int accum, void* out, int ldc,
                  const float* epT = nullptr, const float* epS = nullptr) {
    dim3 grid(N / 128, M / 128);
    gemm_wmma_bf16<<<grid, 256, 0, stream>>>(A, lda, W, ldw, bias, scale, act, K,
                                             outmode, accum, out, ldc, epT, epS);
  };

  // Shared MLP on S and T (reuse hidden buffer)
  gemm(Sb, DIMD, w1b, DIMD, b1, 1.0f, 1, MTOK, HIDD, DIMD, 1, 0, Hb, HIDD);
  gemm(Hb, HIDD, w2b, HIDD, b2, 1.0f, 0, MTOK, DIMD, HIDD, 1, 0, Smb, DIMD);
  gemm(Tb, DIMD, w1b, DIMD, b1, 1.0f, 1, MTOK, HIDD, DIMD, 1, 0, Hb, HIDD);
  gemm(Hb, HIDD, w2b, HIDD, b2, 1.0f, 0, MTOK, DIMD, HIDD, 1, 0, Tmb, DIMD);

  // ss: (S_m,S_m)  st: (T_m,T_m)  cst: (S_m,T_m)  cts: (T_m,S_m)
  const unsigned short* xsrc[4] = { Smb, Tmb, Smb, Tmb }; // -> Q
  const unsigned short* ysrc[4] = { Smb, Tmb, Tmb, Smb }; // -> K,V
  const float invsq = 1.0f / 16.0f;  // 1/sqrt(256)

  for (int att = 0; att < 4; ++att) {
    const int wi = att * 3;
    const float* qbias = (const float*)d_in[9 + 2 * (wi + 0)];
    const float* kbias = (const float*)d_in[9 + 2 * (wi + 1)];
    const float* vbias = (const float*)d_in[9 + 2 * (wi + 2)];
    gemm(xsrc[att], DIMD, awb[wi + 0], DIMD, qbias, 1.0f, 0, MTOK, DIMD, DIMD, 1, 0, Qb,   DIMD);
    gemm(ysrc[att], DIMD, awb[wi + 1], DIMD, kbias, 1.0f, 0, MTOK, DIMD, DIMD, 1, 0, Kbuf, DIMD);
    gemm(ysrc[att], DIMD, awb[wi + 2], DIMD, vbias, 1.0f, 0, MTOK, DIMD, DIMD, 2, 0, Vt,   MTOK); // V^T

    for (int b = 0; b < BATCH; ++b) {
      const unsigned short* Qbb = Qb   + (size_t)b * SEQ * DIMD;
      const unsigned short* Kbb = Kbuf + (size_t)b * SEQ * DIMD;
      // scores = Q K^T / 16
      gemm(Qbb, DIMD, Kbb, DIMD, nullptr, invsq, 0, SEQ, SEQ, DIMD, 0, 0, Sc, SEQ);
      softmax_rows_4096<<<SEQ, 256, 0, stream>>>(Sc, Pb);
      // combined += P @ V   (att 0 initializes)
      gemm(Pb, SEQ, Vt + (size_t)b * SEQ, MTOK, nullptr, 1.0f, 0, SEQ, DIMD, SEQ,
           0, att ? 1 : 0, Comb + (size_t)b * SEQ * DIMD, DIMD);
    }
  }

  cvt(Comb, Combb, MTOK * DIMD);
  // out = (combined @ fc_w^T + fc_b) * T + S
  gemm(Combb, DIMD, fcwb, DIMD, fcb, 1.0f, 0, MTOK, DIMD, DIMD, 3, 0, d_out, DIMD, T, S);
}
